// OldLSSModel_790273982602
// MI455X (gfx1250) — matrix operand order repaired and checked by reading, not compiled, not run on hardware
//
#include <hip/hip_runtime.h>
#include <hip/hip_bf16.h>

typedef _Float16 half_t;
typedef _Float16 v8h  __attribute__((ext_vector_type(8)));
typedef _Float16 v16h __attribute__((ext_vector_type(16)));
typedef float    v8f  __attribute__((ext_vector_type(8)));

#if defined(__HIP_DEVICE_COMPILE__) && __has_builtin(__builtin_amdgcn_global_load_async_to_lds_b128)
#define HAVE_ASYNC_LDS 1
typedef int v4i_vs __attribute__((vector_size(16)));
typedef __attribute__((address_space(1))) v4i_vs gv4i;   // global 16B vector
typedef __attribute__((address_space(3))) v4i_vs lv4i;   // LDS   16B vector
#else
#define HAVE_ASYNC_LDS 0
#endif

// two 16-byte chunks (16 halves) of the B tile: global -> LDS (async DMA when available)
__device__ __forceinline__ void stage_b128(const half_t* src, half_t* dst)
{
#if HAVE_ASYNC_LDS
    __builtin_amdgcn_global_load_async_to_lds_b128((gv4i*)src,       (lv4i*)dst,       0, 0);
    __builtin_amdgcn_global_load_async_to_lds_b128((gv4i*)(src + 8), (lv4i*)(dst + 8), 0, 0);
#else
    *(v8h*)dst       = *(const v8h*)src;
    *(v8h*)(dst + 8) = *(const v8h*)(src + 8);
#endif
}

#define LDS_COL 40   // LDS column stride in halves: 80B -> conflict-free, 16B aligned

// ---------------------------------------------------------------------------
// Implicit-GEMM conv: out[M][CoutP] = im2col(in)[M][K] * wt[K][CoutP] (+bias)
// in : NHWC f16, Cin multiple of 32
// wt : [CoutP][KH][KW][Cin] f16 (pre-packed, zero padded), CoutP % 64 == 0
// Block = 128 threads = 4 waves; WG tile M=64 (4 waves x 16), N=64.
// B tiles (64 cols x 32 K) are double-buffered through LDS with async
// global->LDS DMA (ASYNCcnt) issued one K-step ahead; each wave then runs 4
// independent v_wmma_f32_16x16x32_f16 accumulator chains per K-step.
// ---------------------------------------------------------------------------
__global__ __launch_bounds__(128)
void conv_wmma(const half_t* __restrict__ in, const half_t* __restrict__ wt,
               const float* __restrict__ bias, half_t* __restrict__ out,
               int NB, int IH, int IW, int Cin, int OH, int OW, int CoutP,
               int KH, int KW, int stride, int pad)
{
    __shared__ half_t smem[2][64 * LDS_COL];

    const int tid  = threadIdx.x;
    const int lane = tid & 31;
    const int wave = tid >> 5;
    const int M    = NB * OH * OW;
    const int m0   = blockIdx.x * 64 + wave * 16;
    const int n0   = blockIdx.y * 64;
    const int l16  = lane & 15;
    const int hi   = lane >> 4;

    // A-fragment row handled by this lane (contiguous K, matching B in LDS):
    //   lane<16 holds K {0..15}; lane>=16 holds K {16..31}
    int rowA = m0 + l16;
    bool rvalid = rowA < M;
    int img = 0, oh = 0, ow = 0;
    if (rvalid) {
        img = rowA / (OH * OW);
        int r = rowA - img * (OH * OW);
        oh = r / OW; ow = r - oh * OW;
    }

    // B staging: thread t copies 16 halves of column (n0 + t/2), half t%2
    const size_t kspan = (size_t)(KH * KW) * (size_t)Cin;
    const int scol = tid >> 1;
    const int shc  = tid & 1;
    const half_t* wsrc = wt + (size_t)(n0 + scol) * kspan + shc * 16;
    half_t* sdst0 = &smem[0][scol * LDS_COL + shc * 16];
    half_t* sdst1 = &smem[1][scol * LDS_COL + shc * 16];

    const int nCB = Cin >> 5;            // 32-wide K blocks per (kh,kw)
    const int S   = KH * KW * nCB;       // total K-steps

    v8f acc0 = {}, acc1 = {}, acc2 = {}, acc3 = {};

    int kh = 0, kw = 0, cb = 0;          // current K-step coordinates
    stage_b128(wsrc, sdst0);             // prefetch step 0 into buffer 0

    for (int s = 0; s < S; ++s) {
        const int buf = s & 1;
#if HAVE_ASYNC_LDS
        asm volatile("s_wait_asynccnt 0x0" ::: "memory");   // own wave's DMA done
#endif
        __syncthreads();                  // all waves' staged data visible

        // next K-step coordinates; prefetch into the other buffer
        int cbN = cb + 1, kwN = kw, khN = kh;
        if (cbN == nCB) { cbN = 0; kwN = kw + 1; if (kwN == KW) { kwN = 0; khN = kh + 1; } }
        if (s + 1 < S)
            stage_b128(wsrc + (size_t)(khN * KW + kwN) * Cin + cbN * 32,
                       buf ? sdst0 : sdst1);

        // ---- compute current step -----------------------------------------
        int ih = oh * stride - pad + kh;
        int iw = ow * stride - pad + kw;
        bool v = rvalid && ((unsigned)ih < (unsigned)IH) && ((unsigned)iw < (unsigned)IW);
        const half_t* aptr = in + ((size_t)(img * IH + ih) * IW + iw) * Cin
                                + cb * 32 + hi * 16;
        v8h alo = {}, ahi2 = {};
        if (v) {
            alo  = *(const v8h*)(aptr);        // K hi*16 + 0..7
            ahi2 = *(const v8h*)(aptr + 8);    // K hi*16 + 8..15
        }
        v16h a = __builtin_shufflevector(alo, ahi2,
                  0,1,2,3,4,5,6,7,8,9,10,11,12,13,14,15);

        const half_t* sb = &smem[buf][l16 * LDS_COL + hi * 16];
        v16h b0 = *(const v16h*)(sb + 0 * 16 * LDS_COL);
        v16h b1 = *(const v16h*)(sb + 1 * 16 * LDS_COL);
        v16h b2 = *(const v16h*)(sb + 2 * 16 * LDS_COL);
        v16h b3 = *(const v16h*)(sb + 3 * 16 * LDS_COL);
        acc0 = __builtin_amdgcn_wmma_f32_16x16x32_f16(false, a, false, b0, (short)0, acc0, false, false);
        acc1 = __builtin_amdgcn_wmma_f32_16x16x32_f16(false, a, false, b1, (short)0, acc1, false, false);
        acc2 = __builtin_amdgcn_wmma_f32_16x16x32_f16(false, a, false, b2, (short)0, acc2, false, false);
        acc3 = __builtin_amdgcn_wmma_f32_16x16x32_f16(false, a, false, b3, (short)0, acc3, false, false);

        __syncthreads();                  // readers done before buf is re-staged
        kh = khN; kw = kwN; cb = cbN;
    }

    v8f accs[4] = {acc0, acc1, acc2, acc3};
    #pragma unroll
    for (int nb = 0; nb < 4; ++nb) {
        int col = n0 + nb * 16 + l16;
        float bval = bias ? bias[col] : 0.0f;
        #pragma unroll
        for (int r = 0; r < 8; ++r) {     // C/D layout: row = r + hi*8, col = l16
            int row = m0 + hi * 8 + r;
            if (row < M)
                out[(size_t)row * CoutP + col] = (half_t)(accs[nb][r] + bval);
        }
    }
}

// ---------------------------------------------------------------------------
// conv1: 7x7 s2 p3, Cin=3 (channel count too small for WMMA) — direct VALU.
// imgs NCHW f32 [6][3][256][704] -> raw NHWC f16 [6*128*352][64]
// ---------------------------------------------------------------------------
__global__ void conv1_direct(const float* __restrict__ img,
                             const float* __restrict__ w,
                             half_t* __restrict__ out)
{
    int idx = blockIdx.x * blockDim.x + threadIdx.x;
    const int M = 6 * 128 * 352;
    if (idx >= M * 64) return;
    int c = idx & 63;
    int m = idx >> 6;
    int n = m / (128 * 352);
    int r = m - n * (128 * 352);
    int oh = r / 352, ow = r % 352;
    float acc = 0.0f;
    for (int ci = 0; ci < 3; ++ci) {
        const float* ip = img + (size_t)(n * 3 + ci) * 256 * 704;
        const float* wp = w + (size_t)(c * 3 + ci) * 49;
        for (int kh = 0; kh < 7; ++kh) {
            int ih = oh * 2 - 3 + kh;
            if ((unsigned)ih >= 256u) continue;
            for (int kw = 0; kw < 7; ++kw) {
                int iw = ow * 2 - 3 + kw;
                if ((unsigned)iw >= 704u) continue;
                acc += ip[ih * 704 + iw] * wp[kh * 7 + kw];
            }
        }
    }
    out[(size_t)m * 64 + c] = (half_t)acc;
}

// Per-channel sum / sum-of-squares (one block per channel, tree reduce)
__global__ __launch_bounds__(256)
void bn_stats(const half_t* __restrict__ x, float* __restrict__ stats,
              int Npix, int C)
{
    __shared__ float ss[256], sq[256];
    int c = blockIdx.x, t = threadIdx.x;
    float s = 0.0f, s2 = 0.0f;
    for (int i = t; i < Npix; i += 256) {
        float v = (float)x[(size_t)i * C + c];
        s += v; s2 += v * v;
    }
    ss[t] = s; sq[t] = s2; __syncthreads();
    for (int o = 128; o > 0; o >>= 1) {
        if (t < o) { ss[t] += ss[t + o]; sq[t] += sq[t + o]; }
        __syncthreads();
    }
    if (t == 0) { stats[c] = ss[0]; stats[C + c] = sq[0]; }
}

// y = relu?( g*(x-mean)*rsqrt(var+eps)+b  (+res) )
__global__ void bn_apply(const half_t* __restrict__ x, const float* __restrict__ stats,
                         const float* __restrict__ g, const float* __restrict__ b,
                         const half_t* __restrict__ res, half_t* __restrict__ y,
                         int Npix, int C, int relu)
{
    int idx = blockIdx.x * blockDim.x + threadIdx.x;
    if (idx >= Npix * C) return;
    int c = idx % C;
    float n = (float)Npix;
    float mean = stats[c] / n;
    float var  = stats[C + c] / n - mean * mean;
    float v = g[c] * ((float)x[idx] - mean) * rsqrtf(var + 1e-5f) + b[c];
    if (res)  v += (float)res[idx];
    if (relu) v = fmaxf(v, 0.0f);
    y[idx] = (half_t)v;
}

// 3x3 s2 p1 maxpool: [6][128][352][64] -> [6][64][176][64]
__global__ void maxpool_k(const half_t* __restrict__ in, half_t* __restrict__ out)
{
    int idx = blockIdx.x * blockDim.x + threadIdx.x;
    const int M = 6 * 64 * 176;
    if (idx >= M * 64) return;
    int c = idx & 63;
    int m = idx >> 6;
    int n = m / (64 * 176);
    int r = m - n * (64 * 176);
    int oh = r / 176, ow = r % 176;
    float best = -3.4e38f;
    for (int kh = 0; kh < 3; ++kh) {
        int ih = oh * 2 - 1 + kh;
        if ((unsigned)ih >= 128u) continue;
        for (int kw = 0; kw < 3; ++kw) {
            int iw = ow * 2 - 1 + kw;
            if ((unsigned)iw >= 352u) continue;
            float v = (float)in[((size_t)(n * 128 + ih) * 352 + iw) * 64 + c];
            best = fmaxf(best, v);
        }
    }
    out[(size_t)m * 64 + c] = (half_t)best;
}

// OIHW f32 -> [CoutP][KH][KW][Cin] f16 (zero padded output channels)
__global__ void pack_w(const float* __restrict__ w, half_t* __restrict__ wt,
                       int Cout, int Cin, int K, int CoutP)
{
    int idx = blockIdx.x * blockDim.x + threadIdx.x;
    int total = CoutP * K * K * Cin;
    if (idx >= total) return;
    int ci = idx % Cin; int t = idx / Cin;
    int kw = t % K; t /= K;
    int kh = t % K; int co = t / K;
    float v = (co < Cout) ? w[((size_t)(co * Cin + ci) * K + kh) * K + kw] : 0.0f;
    wt[idx] = (half_t)v;
}

__global__ void pack_bias(const float* __restrict__ b, float* __restrict__ bp,
                          int n, int npad)
{
    int i = blockIdx.x * blockDim.x + threadIdx.x;
    if (i < npad) bp[i] = (i < n) ? b[i] : 0.0f;
}

// softmax over 41 depth bins + split context: head [1056][128]
__global__ void head_softmax(const half_t* __restrict__ head,
                             float* __restrict__ depth, float* __restrict__ ctx)
{
    int p = blockIdx.x * blockDim.x + threadIdx.x;
    if (p >= 1056) return;
    const half_t* h = head + (size_t)p * 128;
    float mx = -3.4e38f;
    for (int i = 0; i < 41; ++i) mx = fmaxf(mx, (float)h[i]);
    float s = 0.0f;
    for (int i = 0; i < 41; ++i) s += expf((float)h[i] - mx);
    float inv = 1.0f / s;
    for (int i = 0; i < 41; ++i) depth[p * 41 + i] = expf((float)h[i] - mx) * inv;
    for (int j = 0; j < 64; ++j) ctx[p * 64 + j] = (float)h[41 + j];
}

__global__ void zero_f32(float* __restrict__ p, int n)
{
    int i = blockIdx.x * blockDim.x + threadIdx.x;
    if (i < n) p[i] = 0.0f;
}

// frustum projection + voxel scatter: atomicAdd of depth*ctx into BEV grid
__global__ void splat(const float* __restrict__ depth, const float* __restrict__ ctx,
                      const float* __restrict__ rots, const float* __restrict__ trans,
                      const float* __restrict__ intr, float* __restrict__ bev)
{
    int idx = blockIdx.x * blockDim.x + threadIdx.x;
    if (idx >= 6 * 41 * 176) return;
    int pix = idx % 176; int t = idx / 176;
    int di  = t % 41;    int cam = t / 41;
    int fh = pix / 22, fw = pix % 22;
    float X = fw * (703.0f / 21.0f);
    float Y = fh * (255.0f / 7.0f);
    float D = 4.0f + (float)di;
    const float* K = intr + cam * 9;
    float px = (X - K[2]) * D / K[0];
    float py = (Y - K[5]) * D / K[4];
    const float* R = rots + cam * 9;
    const float* T = trans + cam * 3;
    float gx = R[0] * px + R[1] * py + R[2] * D + T[0];
    float gy = R[3] * px + R[4] * py + R[5] * D + T[1];
    if (gx >= -50.0f && gx < 50.0f && gy >= -50.0f && gy < 50.0f) {
        int cx = (int)floorf((gx + 50.0f) * 2.0f);
        int cy = (int)floorf((gy + 50.0f) * 2.0f);
        float dv = depth[(size_t)(cam * 176 + pix) * 41 + di];
        const float* cp = ctx + (size_t)(cam * 176 + pix) * 64;
        float* bp = bev + ((size_t)cx * 200 + cy) * 64;
        for (int c = 0; c < 64; ++c)
            atomicAdd(bp + c, cp[c] * dv);
    }
}

__global__ void f32_to_f16(const float* __restrict__ in, half_t* __restrict__ out, int n)
{
    int i = blockIdx.x * blockDim.x + threadIdx.x;
    if (i < n) out[i] = (half_t)in[i];
}

// final 1x1 conv 64->1
__global__ void dec_conv(const half_t* __restrict__ act, const float* __restrict__ w,
                         const float* __restrict__ b, float* __restrict__ out)
{
    int p = blockIdx.x * blockDim.x + threadIdx.x;
    if (p >= 40000) return;
    float acc = b[0];
    const half_t* a = act + (size_t)p * 64;
    for (int c = 0; c < 64; ++c) acc += (float)a[c] * w[c];
    out[p] = acc;
}

// ---------------------------------------------------------------------------
extern "C" void kernel_launch(void* const* d_in, const int* in_sizes, int n_in,
                              void* d_out, int out_size, void* d_ws, size_t ws_size,
                              hipStream_t stream)
{
    (void)in_sizes; (void)n_in; (void)out_size; (void)ws_size;
    const float* imgs  = (const float*)d_in[0];
    const float* rots  = (const float*)d_in[1];
    const float* trans = (const float*)d_in[2];
    const float* intr  = (const float*)d_in[3];
    // params (insertion order): 4 conv1, 5 bn1g, 6 bn1b, then per block
    // {c1,g1,b1,c2,g2,b2[,cd,gd,bd]}, then 64 head_w, 65 head_b, 66 bev_w,
    // 67 bev_b, 68 bevbng, 69 bevbnb, 70 dec_w, 71 dec_b.

    size_t off = 0;
    auto alloc = [&](size_t bytes) -> void* {
        void* p = (char*)d_ws + off;
        off += (bytes + 255) & ~(size_t)255;
        return p;
    };

    struct CW { int widx, Cout, Cin, K, CoutP; half_t* wt; };
    CW cw[21] = {
        { 7, 64, 64,3, 64,nullptr}, {10, 64, 64,3, 64,nullptr},
        {13, 64, 64,3, 64,nullptr}, {16, 64, 64,3, 64,nullptr},
        {19,128, 64,3,128,nullptr}, {22,128,128,3,128,nullptr}, {25,128, 64,1,128,nullptr},
        {28,128,128,3,128,nullptr}, {31,128,128,3,128,nullptr},
        {34,256,128,3,256,nullptr}, {37,256,256,3,256,nullptr}, {40,256,128,1,256,nullptr},
        {43,256,256,3,256,nullptr}, {46,256,256,3,256,nullptr},
        {49,512,256,3,512,nullptr}, {52,512,512,3,512,nullptr}, {55,512,256,1,512,nullptr},
        {58,512,512,3,512,nullptr}, {61,512,512,3,512,nullptr},
        {64,105,512,1,128,nullptr},          // head: padded 105 -> 128 (N-tile 64)
        {66, 64, 64,3, 64,nullptr},
    };
    for (int i = 0; i < 21; ++i) {
        int total = cw[i].CoutP * cw[i].K * cw[i].K * cw[i].Cin;
        cw[i].wt = (half_t*)alloc((size_t)total * sizeof(half_t));
        pack_w<<<(total + 255) / 256, 256, 0, stream>>>(
            (const float*)d_in[cw[i].widx], cw[i].wt,
            cw[i].Cout, cw[i].Cin, cw[i].K, cw[i].CoutP);
    }

    const int M1 = 6 * 128 * 352;      // conv1 output pixels
    const int MP = 6 * 64 * 176;       // post-maxpool pixels

    half_t* RAW   = (half_t*)alloc((size_t)M1 * 64 * sizeof(half_t));
    half_t* RAWD  = (half_t*)alloc((size_t)16896 * 128 * sizeof(half_t));
    half_t* ACT0  = (half_t*)alloc((size_t)M1 * 64 * sizeof(half_t));
    half_t* ACT1  = (half_t*)alloc((size_t)MP * 64 * sizeof(half_t));
    half_t* ACT2  = (half_t*)alloc((size_t)MP * 64 * sizeof(half_t));
    half_t* ACT3  = (half_t*)alloc((size_t)MP * 64 * sizeof(half_t));
    half_t* ACT4  = (half_t*)alloc((size_t)MP * 64 * sizeof(half_t));
    float*  STATS  = (float*)alloc(2 * 512 * sizeof(float));
    float*  STATSD = (float*)alloc(2 * 512 * sizeof(float));
    float*  HEADB  = (float*)alloc(128 * sizeof(float));
    float*  DEPTH  = (float*)alloc((size_t)1056 * 41 * sizeof(float));
    float*  CTX    = (float*)alloc((size_t)1056 * 64 * sizeof(float));
    float*  BEV    = (float*)alloc((size_t)40000 * 64 * sizeof(float));
    half_t* BEVH   = (half_t*)alloc((size_t)40000 * 64 * sizeof(half_t));
    half_t* RAWB   = (half_t*)alloc((size_t)40000 * 64 * sizeof(half_t));
    half_t* BEVA   = (half_t*)alloc((size_t)40000 * 64 * sizeof(half_t));

    auto launch_conv = [&](const half_t* in, const CW& c, const float* bias,
                           half_t* out, int NB, int IH, int IW, int OH, int OW,
                           int stride, int pad) {
        int M = NB * OH * OW;
        dim3 g((M + 63) / 64, c.CoutP / 64);
        conv_wmma<<<g, 128, 0, stream>>>(in, c.wt, bias, out, NB, IH, IW, c.Cin,
                                         OH, OW, c.CoutP, c.K, c.K, stride, pad);
    };
    auto launch_stats = [&](const half_t* x, float* st, int Npix, int C) {
        bn_stats<<<C, 256, 0, stream>>>(x, st, Npix, C);
    };
    auto launch_apply = [&](const half_t* x, const float* st, int gi, int bi,
                            const half_t* res, half_t* y, int Npix, int C, int relu) {
        bn_apply<<<((size_t)Npix * C + 255) / 256, 256, 0, stream>>>(
            x, st, (const float*)d_in[gi], (const float*)d_in[bi],
            res, y, Npix, C, relu);
    };

    // ---- stem: conv1 + BN + ReLU + maxpool --------------------------------
    conv1_direct<<<((size_t)M1 * 64 + 255) / 256, 256, 0, stream>>>(
        imgs, (const float*)d_in[4], RAW);
    launch_stats(RAW, STATS, M1, 64);
    launch_apply(RAW, STATS, 5, 6, nullptr, ACT0, M1, 64, 1);
    maxpool_k<<<((size_t)MP * 64 + 255) / 256, 256, 0, stream>>>(ACT0, ACT1);

    // ---- residual stages --------------------------------------------------
    struct BD { int cwi, pbase, has_cd, stride, Cout, IH, IW, OH, OW; };
    const BD blocks[8] = {
        { 0,  7, 0, 1,  64, 64,176, 64,176},
        { 2, 13, 0, 1,  64, 64,176, 64,176},
        { 4, 19, 1, 2, 128, 64,176, 32, 88},
        { 7, 28, 0, 1, 128, 32, 88, 32, 88},
        { 9, 34, 1, 2, 256, 32, 88, 16, 44},
        {12, 43, 0, 1, 256, 16, 44, 16, 44},
        {14, 49, 1, 2, 512, 16, 44,  8, 22},
        {17, 58, 0, 1, 512,  8, 22,  8, 22},
    };
    half_t* cur = ACT1;
    half_t* pong[2] = {ACT3, ACT1};
    for (int bi = 0; bi < 8; ++bi) {
        const BD& B = blocks[bi];
        int Mout = 6 * B.OH * B.OW;
        half_t* outb = pong[bi & 1];
        launch_conv(cur, cw[B.cwi], nullptr, RAW, 6, B.IH, B.IW, B.OH, B.OW, B.stride, 1);
        launch_stats(RAW, STATS, Mout, B.Cout);
        launch_apply(RAW, STATS, B.pbase + 1, B.pbase + 2, nullptr, ACT2, Mout, B.Cout, 1);
        launch_conv(ACT2, cw[B.cwi + 1], nullptr, RAW, 6, B.OH, B.OW, B.OH, B.OW, 1, 1);
        launch_stats(RAW, STATS, Mout, B.Cout);
        const half_t* identity = cur;
        if (B.has_cd) {
            launch_conv(cur, cw[B.cwi + 2], nullptr, RAWD, 6, B.IH, B.IW, B.OH, B.OW, B.stride, 0);
            launch_stats(RAWD, STATSD, Mout, B.Cout);
            launch_apply(RAWD, STATSD, B.pbase + 7, B.pbase + 8, nullptr, ACT4, Mout, B.Cout, 0);
            identity = ACT4;
        }
        launch_apply(RAW, STATS, B.pbase + 4, B.pbase + 5, identity, outb, Mout, B.Cout, 1);
        cur = outb;
    }

    // ---- head: 1x1 conv 512 -> 105 (padded 128) + bias --------------------
    pack_bias<<<1, 128, 0, stream>>>((const float*)d_in[65], HEADB, 105, 128);
    launch_conv(cur, cw[19], HEADB, RAW, 6, 8, 22, 8, 22, 1, 0);
    head_softmax<<<(1056 + 255) / 256, 256, 0, stream>>>(RAW, DEPTH, CTX);

    // ---- splat to BEV -----------------------------------------------------
    zero_f32<<<(40000 * 64 + 255) / 256, 256, 0, stream>>>(BEV, 40000 * 64);
    splat<<<(6 * 41 * 176 + 255) / 256, 256, 0, stream>>>(
        DEPTH, CTX, rots, trans, intr, BEV);

    // ---- BEV head: 3x3 conv + BN + ReLU, then 1x1 decode ------------------
    f32_to_f16<<<(40000 * 64 + 255) / 256, 256, 0, stream>>>(BEV, BEVH, 40000 * 64);
    launch_conv(BEVH, cw[20], (const float*)d_in[67], RAWB, 1, 200, 200, 200, 200, 1, 1);
    launch_stats(RAWB, STATS, 40000, 64);
    launch_apply(RAWB, STATS, 68, 69, nullptr, BEVA, 40000, 64, 1);
    dec_conv<<<(40000 + 255) / 256, 256, 0, stream>>>(
        BEVA, (const float*)d_in[70], (const float*)d_in[71], (float*)d_out);
}
